// vqcodebook_CL_8581344657813
// MI455X (gfx1250) — matrix operand order repaired and checked by reading, not compiled, not run on hardware
//
#include <hip/hip_runtime.h>
#include <hip/hip_bf16.h>
#include <stdint.h>

typedef __attribute__((ext_vector_type(8)))  __bf16 v8bf;
typedef __attribute__((ext_vector_type(16))) __bf16 v16bf;
typedef __attribute__((ext_vector_type(8)))  float  v8f;
typedef __attribute__((ext_vector_type(4)))  int    v4i;

#define N_ROWS  65536
#define DIM     256
#define K_CODES 4096
#define ROWS_PER_BLK 256          // 8 waves * 2 * 16-row WMMA tiles
#define E_CHUNK 64                // codebook rows staged in LDS per step
#define NCHUNK  (K_CODES / E_CHUNK)

#if defined(__has_builtin)
#if __has_builtin(__builtin_amdgcn_global_load_async_to_lds_b128) && \
    __has_builtin(__builtin_amdgcn_s_wait_asynccnt)
#define USE_ASYNC_LDS 1
#endif
#endif

__device__ __forceinline__ unsigned short f32_to_bf16(float f) {
    union { float f; unsigned u; } v; v.f = f;
    unsigned r = v.u + 0x7FFFu + ((v.u >> 16) & 1u);   // round-to-nearest-even
    return (unsigned short)(r >> 16);
}

// ---------------------------------------------------------------------------
// Normalize rows of [nrows, 256] fp32: write bf16 copy, optional fp32 copy,
// optional per-row inverse norm. One wave per row (lane handles 8 elems).
// ---------------------------------------------------------------------------
__global__ void vq_normalize_kernel(const float* __restrict__ in,
                                    unsigned short* __restrict__ out16,
                                    float* __restrict__ out32,
                                    float* __restrict__ inv_out) {
    const int wave = threadIdx.x >> 5, lane = threadIdx.x & 31;
    const int row = blockIdx.x * 8 + wave;
    const float4* src = (const float4*)(in + (size_t)row * DIM);
    float4 a = src[lane * 2 + 0];
    float4 b = src[lane * 2 + 1];
    float s = a.x*a.x + a.y*a.y + a.z*a.z + a.w*a.w
            + b.x*b.x + b.y*b.y + b.z*b.z + b.w*b.w;
    #pragma unroll
    for (int m = 1; m < 32; m <<= 1) s += __shfl_xor(s, m, 32);
    const float inv = 1.0f / fmaxf(sqrtf(s), 1e-12f);
    a.x *= inv; a.y *= inv; a.z *= inv; a.w *= inv;
    b.x *= inv; b.y *= inv; b.z *= inv; b.w *= inv;
    if (out32) {
        float4* d = (float4*)(out32 + (size_t)row * DIM);
        d[lane * 2 + 0] = a; d[lane * 2 + 1] = b;
    }
    unsigned short us[8] = { f32_to_bf16(a.x), f32_to_bf16(a.y), f32_to_bf16(a.z), f32_to_bf16(a.w),
                             f32_to_bf16(b.x), f32_to_bf16(b.y), f32_to_bf16(b.z), f32_to_bf16(b.w) };
    *(uint4*)(out16 + (size_t)row * DIM + lane * 8) = *(const uint4*)us;
    if (inv_out && lane == 0) inv_out[row] = inv;
}

// ---------------------------------------------------------------------------
// Argmax over K of x_n . e_n via v_wmma_f32_16x16x32_bf16.
// Block = 256 thr (8 waves); wave w owns rows [blk*256 + 32w, +32) as two
// register-resident A tiles, so each B fragment from LDS feeds TWO WMMAs.
// Codebook streamed through double-buffered LDS (2 x 32KB) using async
// global->LDS DMA (ASYNCcnt) when available.
// ---------------------------------------------------------------------------
#if USE_ASYNC_LDS
__device__ __forceinline__ void issue_chunk_async(const unsigned short* __restrict__ e16,
                                                  unsigned short* buf, int chunk, int tid) {
    const uint4* src = (const uint4*)(e16 + (size_t)chunk * E_CHUNK * DIM);
    uint4* dst = (uint4*)buf;
    #pragma unroll
    for (int i = 0; i < 8; ++i) {
        __builtin_amdgcn_global_load_async_to_lds_b128(
            (__attribute__((address_space(1))) v4i*)(src + tid + i * 256),
            (__attribute__((address_space(3))) v4i*)(dst + tid + i * 256),
            0, 0);
    }
}
#endif

__global__ void __launch_bounds__(256, 1)
vq_argmax_wmma_kernel(const unsigned short* __restrict__ x16,
                      const unsigned short* __restrict__ e16,
                      int* __restrict__ idx_out,
                      float* __restrict__ out /* d_out base */) {
#if USE_ASYNC_LDS
    __shared__ unsigned short lds_e[2][E_CHUNK * DIM];   // 2 x 32 KB
#else
    __shared__ unsigned short lds_e[1][E_CHUNK * DIM];   // 32 KB
#endif

    const int wave = threadIdx.x >> 5;
    const int lane = threadIdx.x & 31;
    const int h    = lane >> 4;        // half select (ISA bf16 VGPR layout)
    const int l16  = lane & 15;
    const int rowBase = blockIdx.x * ROWS_PER_BLK + wave * 32;

    // Resident A fragments for two 16-row tiles: lane l16 holds the row;
    // elems 0..15 = K kk*32 + h*16 + {0..15}
    v16bf A0[8], A1[8];
    {
        const unsigned short* xr0 = x16 + (size_t)(rowBase + l16) * DIM;
        const unsigned short* xr1 = x16 + (size_t)(rowBase + 16 + l16) * DIM;
        #pragma unroll
        for (int kk = 0; kk < 8; ++kk) {
            v8bf lo0 = *(const v8bf*)(xr0 + kk * 32 + h * 16);
            v8bf hi0 = *(const v8bf*)(xr0 + kk * 32 + h * 16 + 8);
            A0[kk] = __builtin_shufflevector(lo0, hi0, 0,1,2,3,4,5,6,7,8,9,10,11,12,13,14,15);
            v8bf lo1 = *(const v8bf*)(xr1 + kk * 32 + h * 16);
            v8bf hi1 = *(const v8bf*)(xr1 + kk * 32 + h * 16 + 8);
            A1[kk] = __builtin_shufflevector(lo1, hi1, 0,1,2,3,4,5,6,7,8,9,10,11,12,13,14,15);
        }
    }

    float bv0[8], bv1[8];
    int   bi0[8], bi1[8];
    #pragma unroll
    for (int r = 0; r < 8; ++r) {
        bv0[r] = -3.402823e38f; bi0[r] = 0;
        bv1[r] = -3.402823e38f; bi1[r] = 0;
    }

#if USE_ASYNC_LDS
    issue_chunk_async(e16, lds_e[0], 0, threadIdx.x);
#endif

    for (int chunk = 0; chunk < NCHUNK; ++chunk) {
#if USE_ASYNC_LDS
        __builtin_amdgcn_s_wait_asynccnt(0);   // our chunk's DMA landed
        __syncthreads();                        // everyone's landed; old buffer free
        if (chunk + 1 < NCHUNK)
            issue_chunk_async(e16, lds_e[(chunk + 1) & 1], chunk + 1, threadIdx.x);
        const unsigned short* buf = lds_e[chunk & 1];
#else
        __syncthreads();
        {   // synchronous cooperative copy: 64 rows * 256 bf16 = 2048 x uint4
            const uint4* src = (const uint4*)(e16 + (size_t)chunk * E_CHUNK * DIM);
            uint4* dst = (uint4*)lds_e[0];
            #pragma unroll
            for (int i = 0; i < 8; ++i)
                dst[threadIdx.x + i * 256] = src[threadIdx.x + i * 256];
        }
        __syncthreads();
        const unsigned short* buf = lds_e[0];
#endif

        #pragma unroll
        for (int nt = 0; nt < E_CHUNK / 16; ++nt) {
            v8f c0 = {}, c1 = {};
            // B frag: lane l16 holds codebook row (nt*16+l16); lower half K=kk*32+0..15,
            // upper half K=kk*32+16..31 -> one contiguous 32B chunk per lane.
            const unsigned short* erow = buf + (size_t)(nt * 16 + l16) * DIM;
            #pragma unroll
            for (int kk = 0; kk < 8; ++kk) {
                v16bf B = *(const v16bf*)(erow + kk * 32 + h * 16);
                c0 = __builtin_amdgcn_wmma_f32_16x16x32_bf16(
                         false, A0[kk], false, B, (short)0, c0, false, false);
                c1 = __builtin_amdgcn_wmma_f32_16x16x32_bf16(
                         false, A1[kk], false, B, (short)0, c1, false, false);
            }
            const int ncol = chunk * E_CHUNK + nt * 16 + l16;
            #pragma unroll
            for (int r = 0; r < 8; ++r) {
                if (c0[r] > bv0[r]) { bv0[r] = c0[r]; bi0[r] = ncol; }
                if (c1[r] > bv1[r]) { bv1[r] = c1[r]; bi1[r] = ncol; }
            }
        }
    }

    // reduce across the 16 lanes of each half (rows r+8h live in one half);
    // tie-break on lowest index to match argmin semantics.
    #pragma unroll
    for (int m = 1; m < 16; m <<= 1) {
        #pragma unroll
        for (int r = 0; r < 8; ++r) {
            float ov0 = __shfl_xor(bv0[r], m, 32);
            int   oi0 = __shfl_xor(bi0[r], m, 32);
            if (ov0 > bv0[r] || (ov0 == bv0[r] && oi0 < bi0[r])) { bv0[r] = ov0; bi0[r] = oi0; }
            float ov1 = __shfl_xor(bv1[r], m, 32);
            int   oi1 = __shfl_xor(bi1[r], m, 32);
            if (ov1 > bv1[r] || (ov1 == bv1[r] && oi1 < bi1[r])) { bv1[r] = ov1; bi1[r] = oi1; }
        }
    }
    if (l16 == 0) {   // lanes 0 and 16 hold results for rows r and r+8
        #pragma unroll
        for (int r = 0; r < 8; ++r) {
            const int row0 = rowBase + 8 * h + r;
            idx_out[row0] = bi0[r];
            out[(size_t)N_ROWS * DIM + 1 + row0] = (float)bi0[r];
            const int row1 = rowBase + 16 + 8 * h + r;
            idx_out[row1] = bi1[r];
            out[(size_t)N_ROWS * DIM + 1 + row1] = (float)bi1[r];
        }
    }
}

// ---------------------------------------------------------------------------
// Gather fp32 normalized codebook rows -> output; per-block SSE partials.
// ---------------------------------------------------------------------------
__global__ void vq_gather_loss_kernel(const float* __restrict__ x,
                                      const float* __restrict__ inv_x,
                                      const int* __restrict__ idx,
                                      const float* __restrict__ e32,
                                      float* __restrict__ out_q,
                                      float* __restrict__ partials) {
    __shared__ float sacc[8];
    const int wave = threadIdx.x >> 5, lane = threadIdx.x & 31;
    const int row = blockIdx.x * 8 + wave;
    const int k = idx[row];
    const float4* qv = (const float4*)(e32 + (size_t)k * DIM);
    const float4* xv = (const float4*)(x + (size_t)row * DIM);
    const float inv = inv_x[row];
    float4 q0 = qv[lane * 2 + 0], q1 = qv[lane * 2 + 1];
    float4 x0 = xv[lane * 2 + 0], x1 = xv[lane * 2 + 1];
    float acc = 0.0f;
    float d;
    d = q0.x - x0.x * inv; acc += d * d;
    d = q0.y - x0.y * inv; acc += d * d;
    d = q0.z - x0.z * inv; acc += d * d;
    d = q0.w - x0.w * inv; acc += d * d;
    d = q1.x - x1.x * inv; acc += d * d;
    d = q1.y - x1.y * inv; acc += d * d;
    d = q1.z - x1.z * inv; acc += d * d;
    d = q1.w - x1.w * inv; acc += d * d;
    float4* o = (float4*)(out_q + (size_t)row * DIM);
    o[lane * 2 + 0] = q0; o[lane * 2 + 1] = q1;
    #pragma unroll
    for (int m = 1; m < 32; m <<= 1) acc += __shfl_xor(acc, m, 32);
    if (lane == 0) sacc[wave] = acc;
    __syncthreads();
    if (threadIdx.x == 0) {
        float t = 0.0f;
        #pragma unroll
        for (int i = 0; i < 8; ++i) t += sacc[i];
        partials[blockIdx.x] = t;
    }
}

__global__ void vq_loss_reduce_kernel(const float* __restrict__ partials,
                                      float* __restrict__ loss_out, int n) {
    __shared__ float s[8];
    float acc = 0.0f;
    for (int i = threadIdx.x; i < n; i += 256) acc += partials[i];
    #pragma unroll
    for (int m = 1; m < 32; m <<= 1) acc += __shfl_xor(acc, m, 32);
    const int wave = threadIdx.x >> 5, lane = threadIdx.x & 31;
    if (lane == 0) s[wave] = acc;
    __syncthreads();
    if (threadIdx.x == 0) {
        float t = 0.0f;
        #pragma unroll
        for (int i = 0; i < 8; ++i) t += s[i];
        // loss = q_latent + 0.25*e_latent = 1.25 * mean((q - x_n)^2)
        *loss_out = t * (1.25f / ((float)N_ROWS * (float)DIM));
    }
}

// ---------------------------------------------------------------------------
extern "C" void kernel_launch(void* const* d_in, const int* in_sizes, int n_in,
                              void* d_out, int out_size, void* d_ws, size_t ws_size,
                              hipStream_t stream) {
    const float* x  = (const float*)d_in[0];   // [65536, 256]
    const float* ew = (const float*)d_in[1];   // [4096, 256]
    float* out = (float*)d_out;                // [N*D] quantized | [1] loss | [N] indices

    char* w = (char*)d_ws;
    size_t off = 0;
    unsigned short* x16 = (unsigned short*)(w + off); off += (size_t)N_ROWS * DIM * 2;  // 32 MB
    unsigned short* e16 = (unsigned short*)(w + off); off += (size_t)K_CODES * DIM * 2; //  2 MB
    float* e32   = (float*)(w + off); off += (size_t)K_CODES * DIM * 4;                 //  4 MB
    float* inv_x = (float*)(w + off); off += (size_t)N_ROWS * 4;
    int*   idxws = (int*)(w + off);   off += (size_t)N_ROWS * 4;
    float* parts = (float*)(w + off); off += (size_t)(N_ROWS / 8) * 4;

    // 1) normalize x -> bf16 + inv norms
    vq_normalize_kernel<<<N_ROWS / 8, 256, 0, stream>>>(x, x16, nullptr, inv_x);
    // 2) normalize codebook -> bf16 + fp32
    vq_normalize_kernel<<<K_CODES / 8, 256, 0, stream>>>(ew, e16, e32, nullptr);
    // 3) WMMA similarity argmax (writes indices to ws + float indices to d_out)
    vq_argmax_wmma_kernel<<<N_ROWS / ROWS_PER_BLK, 256, 0, stream>>>(x16, e16, idxws, out);
    // 4) gather quantized rows + per-block loss partials
    vq_gather_loss_kernel<<<N_ROWS / 8, 256, 0, stream>>>(x, inv_x, idxws, e32, out, parts);
    // 5) deterministic final loss reduction
    vq_loss_reduce_kernel<<<1, 256, 0, stream>>>(parts, out + (size_t)N_ROWS * DIM, N_ROWS / 8);
}
